// _Experts_25872882991284
// MI455X (gfx1250) — compile-verified, matrix operand order checked by reading, not run on hardware
//
#include <hip/hip_runtime.h>

#define TT 8192
#define HH 2048
#define II 1408
#define EE 16
#define TOPK 2

#define BM 64
#define BN 64
#define BK 64
#define LDT 72   // LDS row stride in ushorts (144B, 16B-aligned, pads banks)

typedef __attribute__((ext_vector_type(16))) __bf16 v16bf;
typedef __attribute__((ext_vector_type(8)))  __bf16 v8bf;
typedef __attribute__((ext_vector_type(8)))  float  v8f;
typedef __attribute__((ext_vector_type(4)))  int    v4i;

#if defined(__has_builtin)
#  if __has_builtin(__builtin_amdgcn_global_load_async_to_lds_b128)
#    define HAVE_ASYNC_LDS 1
#  endif
#  if __has_builtin(__builtin_amdgcn_s_wait_asynccnt)
#    define HAVE_WAIT_ASYNC_BUILTIN 1
#  endif
#endif

typedef __attribute__((address_space(1))) v4i gv4i;   // global
typedef __attribute__((address_space(3))) v4i lv4i;   // LDS

__device__ __forceinline__ unsigned short f2bf(float f) {
  unsigned int u = __float_as_uint(f);
  u += 0x7FFFu + ((u >> 16) & 1u);          // round-to-nearest-even
  return (unsigned short)(u >> 16);
}

// 16-byte global -> LDS copy. Prefer CDNA5 async DMA path (ASYNCcnt, no
// destination VGPRs, frees VALU); fall back to a VGPR round-trip copy.
__device__ __forceinline__ void lds_copy16(const unsigned short* g, unsigned short* l) {
#if defined(HAVE_ASYNC_LDS)
  __builtin_amdgcn_global_load_async_to_lds_b128((gv4i*)g, (lv4i*)l, 0, 0);
#else
  *(uint4*)l = *(const uint4*)g;
#endif
}

__device__ __forceinline__ void wait_async_and_sync() {
#if defined(HAVE_ASYNC_LDS)
#  if defined(HAVE_WAIT_ASYNC_BUILTIN)
  __builtin_amdgcn_s_wait_asynccnt(0);
#  else
  asm volatile("s_wait_asynccnt 0x0" ::: "memory");
#  endif
#endif
  __syncthreads();
}

// Fragment load per CDNA5 WMMA 16-bit layout (ISA 7.12.2):
// lane 0-15 : row = lidx, K = kb..kb+7 and kb+16..kb+23   (kb = kk)
// lane 16-31: row = lidx, K offsets shifted by +8          (kb = kk+8)
__device__ __forceinline__ v16bf ld_frag(const unsigned short* t, int row, int kb) {
  union { v16bf v; v8bf h[2]; } u;
  u.h[0] = *(const v8bf*)(t + row * LDT + kb);
  u.h[1] = *(const v8bf*)(t + row * LDT + kb + 16);
  return u.v;
}

__global__ void zero_kernel(float* __restrict__ out, float* __restrict__ combine) {
  size_t i = (size_t)blockIdx.x * blockDim.x + threadIdx.x;
  if (i < (size_t)TT * HH) out[i] = 0.0f;
  if (i < (size_t)EE * TT) combine[i] = 0.0f;
}

__global__ void combine_kernel(const int* __restrict__ idx, const float* __restrict__ w,
                               float* __restrict__ combine) {
  int i = blockIdx.x * blockDim.x + threadIdx.x;
  if (i < TT * TOPK) {
    int t = i / TOPK;
    int e = idx[i];
    atomicAdd(combine + (size_t)e * TT + t, w[i]);
  }
}

__global__ void cvtx_kernel(const float* __restrict__ X, unsigned short* __restrict__ Xb) {
  size_t i = ((size_t)blockIdx.x * blockDim.x + threadIdx.x) * 4;
  if (i < (size_t)TT * HH) {
    float4 v = *(const float4*)(X + i);
    unsigned int lo = (unsigned int)f2bf(v.x) | ((unsigned int)f2bf(v.y) << 16);
    unsigned int hi = (unsigned int)f2bf(v.z) | ((unsigned int)f2bf(v.w) << 16);
    uint2 p; p.x = lo; p.y = hi;
    *(uint2*)(Xb + i) = p;
  }
}

// GEMM1 + SwiGLU: h[m0:m0+64, n0:n0+64] = silu(X@Wg^T) * (X@Wu^T), bf16 out
__global__ __launch_bounds__(128) void gemm1_kernel(
    const unsigned short* __restrict__ Xb,   // [T][H] bf16
    const float* __restrict__ Wgu,           // [2I][H] fp32, this expert
    unsigned short* __restrict__ Hb)         // [T][I] bf16
{
  __shared__ unsigned short sX[BM * LDT];
  __shared__ unsigned short sG[BN * LDT];
  __shared__ unsigned short sU[BN * LDT];

  const int n0   = blockIdx.x * BN;
  const int m0   = blockIdx.y * BM;
  const int tid  = threadIdx.x;
  const int lane = tid & 31;
  const int wave = tid >> 5;
  const int wm   = (wave & 1) * 32;
  const int wn   = (wave >> 1) * 32;
  const int lhalf = (lane >> 4) * 8;
  const int lidx  = lane & 15;

  v8f accG[2][2], accU[2][2];
#pragma unroll
  for (int i = 0; i < 2; ++i)
#pragma unroll
    for (int j = 0; j < 2; ++j)
#pragma unroll
      for (int r = 0; r < 8; ++r) { accG[i][j][r] = 0.0f; accU[i][j][r] = 0.0f; }

  for (int k0 = 0; k0 < HH; k0 += BK) {
    // Stage X tile (bf16): async global->LDS, 16B per lane-chunk
#pragma unroll
    for (int it = 0; it < 4; ++it) {
      int i = tid + it * 128;
      int r = i >> 3, c = (i & 7) * 8;
      lds_copy16(Xb + (size_t)(m0 + r) * HH + k0 + c, sX + r * LDT + c);
    }
    // Stage gate & up weight tiles, fp32 -> bf16 on the fly (VALU path)
#pragma unroll
    for (int it = 0; it < 8; ++it) {
      int i = tid + it * 128;
      int r = i >> 4, c = (i & 15) * 4;
      const float* pg = Wgu + (size_t)(n0 + r) * HH + k0 + c;
      float4 g = *(const float4*)pg;
      float4 u = *(const float4*)(pg + (size_t)II * HH);
      uint2 gp, up;
      gp.x = (unsigned int)f2bf(g.x) | ((unsigned int)f2bf(g.y) << 16);
      gp.y = (unsigned int)f2bf(g.z) | ((unsigned int)f2bf(g.w) << 16);
      up.x = (unsigned int)f2bf(u.x) | ((unsigned int)f2bf(u.y) << 16);
      up.y = (unsigned int)f2bf(u.z) | ((unsigned int)f2bf(u.w) << 16);
      *(uint2*)(sG + r * LDT + c) = gp;
      *(uint2*)(sU + r * LDT + c) = up;
    }
    // Prefetch next weight K-slab while computing this one
    if (k0 + BK < HH) {
      const float* np = Wgu + (size_t)(n0 + (tid >> 1)) * HH + (k0 + BK) + (tid & 1) * 32;
      __builtin_prefetch(np, 0, 0);
      __builtin_prefetch(np + (size_t)II * HH, 0, 0);
    }
    wait_async_and_sync();

#pragma unroll
    for (int kk = 0; kk < BK; kk += 32) {
      v16bf a[2], bg[2], bu[2];
#pragma unroll
      for (int i = 0; i < 2; ++i)
        a[i] = ld_frag(sX, wm + i * 16 + lidx, kk + lhalf);
#pragma unroll
      for (int j = 0; j < 2; ++j) {
        bg[j] = ld_frag(sG, wn + j * 16 + lidx, kk + lhalf);
        bu[j] = ld_frag(sU, wn + j * 16 + lidx, kk + lhalf);
      }
#pragma unroll
      for (int i = 0; i < 2; ++i)
#pragma unroll
        for (int j = 0; j < 2; ++j) {
          accG[i][j] = __builtin_amdgcn_wmma_f32_16x16x32_bf16(
              false, a[i], false, bg[j], (short)0, accG[i][j], false, false);
          accU[i][j] = __builtin_amdgcn_wmma_f32_16x16x32_bf16(
              false, a[i], false, bu[j], (short)0, accU[i][j], false, false);
        }
    }
    __syncthreads();
  }

  // Epilogue: SwiGLU, write bf16 h. C layout: VGPR r -> M = lhalf + r, N = lidx
#pragma unroll
  for (int i = 0; i < 2; ++i)
#pragma unroll
    for (int j = 0; j < 2; ++j)
#pragma unroll
      for (int r = 0; r < 8; ++r) {
        int row = m0 + wm + i * 16 + lhalf + r;
        int col = n0 + wn + j * 16 + lidx;
        float g = accG[i][j][r];
        float u = accU[i][j][r];
        float hv = (g / (1.0f + __expf(-g))) * u;
        Hb[(size_t)row * II + col] = f2bf(hv);
      }
}

// GEMM2: out[m, n] += (h @ Wd^T)[m, n] * combine[m]
__global__ __launch_bounds__(128) void gemm2_kernel(
    const unsigned short* __restrict__ Hb,   // [T][I] bf16
    const float* __restrict__ Wd,            // [H][I] fp32, this expert
    const float* __restrict__ cw,            // [T] combine weights, this expert
    float* __restrict__ out)                 // [T][H] fp32
{
  __shared__ unsigned short sA[BM * LDT];
  __shared__ unsigned short sB[BN * LDT];

  const int n0   = blockIdx.x * BN;
  const int m0   = blockIdx.y * BM;
  const int tid  = threadIdx.x;
  const int lane = tid & 31;
  const int wave = tid >> 5;
  const int wm   = (wave & 1) * 32;
  const int wn   = (wave >> 1) * 32;
  const int lhalf = (lane >> 4) * 8;
  const int lidx  = lane & 15;

  v8f acc[2][2];
#pragma unroll
  for (int i = 0; i < 2; ++i)
#pragma unroll
    for (int j = 0; j < 2; ++j)
#pragma unroll
      for (int r = 0; r < 8; ++r) acc[i][j][r] = 0.0f;

  for (int k0 = 0; k0 < II; k0 += BK) {
    // A tile (h, already bf16): async global->LDS
#pragma unroll
    for (int it = 0; it < 4; ++it) {
      int i = tid + it * 128;
      int r = i >> 3, c = (i & 7) * 8;
      lds_copy16(Hb + (size_t)(m0 + r) * II + k0 + c, sA + r * LDT + c);
    }
    // B tile (down-proj weights): fp32 -> bf16 VALU path
#pragma unroll
    for (int it = 0; it < 8; ++it) {
      int i = tid + it * 128;
      int r = i >> 4, c = (i & 15) * 4;
      float4 w4 = *(const float4*)(Wd + (size_t)(n0 + r) * II + k0 + c);
      uint2 p;
      p.x = (unsigned int)f2bf(w4.x) | ((unsigned int)f2bf(w4.y) << 16);
      p.y = (unsigned int)f2bf(w4.z) | ((unsigned int)f2bf(w4.w) << 16);
      *(uint2*)(sB + r * LDT + c) = p;
    }
    if (k0 + BK < II) {
      __builtin_prefetch(Wd + (size_t)(n0 + (tid >> 1)) * II + (k0 + BK) + (tid & 1) * 32, 0, 0);
    }
    wait_async_and_sync();

#pragma unroll
    for (int kk = 0; kk < BK; kk += 32) {
      v16bf a[2], b[2];
#pragma unroll
      for (int i = 0; i < 2; ++i)
        a[i] = ld_frag(sA, wm + i * 16 + lidx, kk + lhalf);
#pragma unroll
      for (int j = 0; j < 2; ++j)
        b[j] = ld_frag(sB, wn + j * 16 + lidx, kk + lhalf);
#pragma unroll
      for (int i = 0; i < 2; ++i)
#pragma unroll
        for (int j = 0; j < 2; ++j)
          acc[i][j] = __builtin_amdgcn_wmma_f32_16x16x32_bf16(
              false, a[i], false, b[j], (short)0, acc[i][j], false, false);
    }
    __syncthreads();
  }

#pragma unroll
  for (int i = 0; i < 2; ++i)
#pragma unroll
    for (int j = 0; j < 2; ++j)
#pragma unroll
      for (int r = 0; r < 8; ++r) {
        int row = m0 + wm + i * 16 + lhalf + r;
        int col = n0 + wn + j * 16 + lidx;
        float c = cw[row];
        float* po = out + (size_t)row * HH + col;
        *po += acc[i][j][r] * c;   // race-free: experts launched sequentially
      }
}

extern "C" void kernel_launch(void* const* d_in, const int* in_sizes, int n_in,
                              void* d_out, int out_size, void* d_ws, size_t ws_size,
                              hipStream_t stream) {
  (void)in_sizes; (void)n_in; (void)out_size; (void)ws_size;
  const float* X    = (const float*)d_in[0];     // [T,H]
  const int*   idx  = (const int*)d_in[1];       // [T,K]
  const float* wts  = (const float*)d_in[2];     // [T,K]
  const float* Wgu  = (const float*)d_in[3];     // [E,2I,H]
  const float* Wd   = (const float*)d_in[4];     // [E,H,I]
  float* out = (float*)d_out;                    // [T,H]

  char* ws = (char*)d_ws;
  float* combine      = (float*)ws;                                           // E*T f32 (512 KB)
  unsigned short* Hb  = (unsigned short*)(ws + (size_t)EE * TT * 4);          // T*I bf16 (23 MB)
  unsigned short* Xb  = (unsigned short*)(ws + (size_t)EE * TT * 4
                                             + (size_t)TT * II * 2);          // T*H bf16 (33.5 MB)

  {
    size_t n = (size_t)TT * HH;
    int blocks = (int)((n + 255) / 256);
    zero_kernel<<<blocks, 256, 0, stream>>>(out, combine);
  }
  combine_kernel<<<(TT * TOPK + 255) / 256, 256, 0, stream>>>(idx, wts, combine);
  cvtx_kernel<<<(int)(((size_t)TT * HH / 4 + 255) / 256), 256, 0, stream>>>(X, Xb);

  dim3 g1(II / BN, TT / BM);   // (22, 128)
  dim3 g2(HH / BN, TT / BM);   // (32, 128)
  for (int e = 0; e < EE; ++e) {
    gemm1_kernel<<<g1, 128, 0, stream>>>(
        Xb, Wgu + (size_t)e * 2 * II * HH, Hb);
    gemm2_kernel<<<g2, 128, 0, stream>>>(
        Hb, Wd + (size_t)e * HH * II, combine + (size_t)e * TT, out);
  }
}
